// RelativeMultiHeadAttention_841813590083
// MI455X (gfx1250) — matrix operand order repaired and checked by reading, not compiled, not run on hardware
//
#include <hip/hip_runtime.h>
#include <hip/hip_bf16.h>

// ---------------------------------------------------------------------------
// Relative Multi-Head Attention (Transformer-XL) for MI455X / gfx1250.
// All GEMMs on v_wmma_f32_16x16x32_bf16 (wave32), double-buffered K-loops,
// 32-bit offsets for saddr+voffset addressing, flash softmax, rel-shift as
// an index remap on PS2 = (q+v_bias) @ p^T.
// ---------------------------------------------------------------------------

typedef __attribute__((ext_vector_type(16))) __bf16 v16bf;
typedef __attribute__((ext_vector_type(8)))  __bf16 v8bf;
typedef __attribute__((ext_vector_type(8)))  float  v8f;
typedef __attribute__((ext_vector_type(4)))  float  f32x4;

#define SLEN  1024
#define DMOD  1024
#define NHEAD 16
#define DHEAD 64
#define BATCH 4
#define MTOK  4096   // BATCH * SLEN

// ---------------- workspace layout (bytes) ----------------
static constexpr size_t WBYTES  = (size_t)DMOD * DMOD * 2;     // 2 MB per weight (bf16)
static constexpr size_t ABYTES  = (size_t)MTOK * DMOD * 2;     // 8 MB per activation (bf16)
static constexpr size_t OFF_WQT = 0;
static constexpr size_t OFF_WKT = OFF_WQT + WBYTES;
static constexpr size_t OFF_WVT = OFF_WKT + WBYTES;
static constexpr size_t OFF_WPT = OFF_WVT + WBYTES;
static constexpr size_t OFF_WOT = OFF_WPT + WBYTES;
static constexpr size_t OFF_QU  = OFF_WOT + WBYTES;
static constexpr size_t OFF_QV  = OFF_QU + ABYTES;
static constexpr size_t OFF_K   = OFF_QV + ABYTES;
static constexpr size_t OFF_VT  = OFF_K  + ABYTES;
static constexpr size_t OFF_P   = OFF_VT + ABYTES;
static constexpr size_t OFF_PS2 = OFF_P  + ABYTES;                       // 128 MB
static constexpr size_t OFF_CTX = OFF_PS2 + (size_t)BATCH * NHEAD * SLEN * SLEN * 2;
// total ~186 MB of d_ws

// ---------------- helpers ----------------
__device__ __forceinline__ v8f vzero8() {
  v8f z = {0.f, 0.f, 0.f, 0.f, 0.f, 0.f, 0.f, 0.f};
  return z;
}

__device__ __forceinline__ v8f wmma_bf16(v16bf a, v16bf b, v8f c) {
  return __builtin_amdgcn_wmma_f32_16x16x32_bf16(false, a, false, b, (short)0, c,
                                                 false, false);
}

// merge the two 16B halves of an A-fragment
__device__ __forceinline__ v16bf mk16(v8bf lo, v8bf hi) {
  v16bf r;
#pragma unroll
  for (int i = 0; i < 8; ++i) { r[i] = lo[i]; r[i + 8] = hi[i]; }
  return r;
}

// 16 f32 (4x f32x4) -> 16 bf16
__device__ __forceinline__ v16bf cvt16(const f32x4 q[4]) {
  v16bf r;
#pragma unroll
  for (int g = 0; g < 4; ++g)
#pragma unroll
    for (int i = 0; i < 4; ++i) r[g * 4 + i] = (__bf16)q[g][i];
  return r;
}

// ---------------------------------------------------------------------------
// K0: weight transpose + f32->bf16 convert (LDS tiled), WT[c][k] = W[k][c]
// ---------------------------------------------------------------------------
__global__ __launch_bounds__(256) void wprep_kernel(const float* __restrict__ W,
                                                    __bf16* __restrict__ WT) {
  __shared__ float tile[32][33];
  const int tx = threadIdx.x, ty = threadIdx.y;
  const unsigned sx = blockIdx.x * 32 + tx;  // source col (c)
  const unsigned sy = blockIdx.y * 32;       // source row base (k)
#pragma unroll
  for (int j = 0; j < 32; j += 8)
    tile[ty + j][tx] = W[(sy + ty + j) * DMOD + sx];
  __syncthreads();
  const unsigned oc = blockIdx.x * 32;       // out row (c)
  const unsigned ok = blockIdx.y * 32 + tx;  // out col (k)
#pragma unroll
  for (int j = 0; j < 32; j += 8)
    WT[(oc + ty + j) * DMOD + ok] = (__bf16)tile[tx][ty + j];
}

// ---------------------------------------------------------------------------
// K1a: projection GEMM, orientation M=c (feature), N=s (token).
// O[s][c] = X[s][:] . WT[c][:] + bias.  Output scattered per-head (b,h,s,d).
// MODE 0 = Q (dual out: +u_bias and +v_bias), 1 = K, 3 = P (bias may be null).
// Double-buffered K loop.
// ---------------------------------------------------------------------------
template <int MODE>
__global__ __launch_bounds__(256) void proj_cs_kernel(
    const float* __restrict__ X, const __bf16* __restrict__ WT,
    const float* __restrict__ bias, const float* __restrict__ ub,
    const float* __restrict__ vb, __bf16* __restrict__ out1,
    __bf16* __restrict__ out2) {
  const int lane = threadIdx.x & 31, wave = threadIdx.x >> 5;
  const int ln = lane & 15, lh = lane >> 4;
  const int c0 = blockIdx.x * 128 + (wave & 1) * 64;
  const int s0 = blockIdx.y * 128 + (wave >> 1) * 32;

  unsigned arow[4], brow[2];
#pragma unroll
  for (int mt = 0; mt < 4; ++mt)
    arow[mt] = (unsigned)(c0 + mt * 16 + ln) * DMOD + lh * 16;
#pragma unroll
  for (int nt = 0; nt < 2; ++nt)
    brow[nt] = (unsigned)(s0 + nt * 16 + ln) * DMOD + lh * 16;

  v8f acc[4][2];
#pragma unroll
  for (int mt = 0; mt < 4; ++mt)
#pragma unroll
    for (int nt = 0; nt < 2; ++nt) acc[mt][nt] = vzero8();

  auto ldA = [&](unsigned kk, v8bf (&dst)[4][2]) {
#pragma unroll
    for (int mt = 0; mt < 4; ++mt) {
      const __bf16* p = WT + (arow[mt] + kk);
      dst[mt][0] = *(const v8bf*)p;
      dst[mt][1] = *(const v8bf*)(p + 8);
    }
  };
  auto ldB = [&](unsigned kk, f32x4 (&dst)[2][4]) {
#pragma unroll
    for (int nt = 0; nt < 2; ++nt) {
      const float* p = X + (brow[nt] + kk);
#pragma unroll
      for (int g = 0; g < 4; ++g) dst[nt][g] = *(const f32x4*)(p + g * 4);
    }
  };
  auto compute = [&](v8bf (&ar)[4][2], f32x4 (&br)[2][4]) {
    v16bf a[4], b[2];
#pragma unroll
    for (int mt = 0; mt < 4; ++mt) a[mt] = mk16(ar[mt][0], ar[mt][1]);
#pragma unroll
    for (int nt = 0; nt < 2; ++nt) b[nt] = cvt16(br[nt]);
#pragma unroll
    for (int mt = 0; mt < 4; ++mt)
#pragma unroll
      for (int nt = 0; nt < 2; ++nt)
        acc[mt][nt] = wmma_bf16(a[mt], b[nt], acc[mt][nt]);
  };

  v8bf araw[2][4][2];
  f32x4 braw[2][2][4];
  ldA(0, araw[0]);
  ldB(0, braw[0]);
  int pb = 0;
#pragma unroll 2
  for (unsigned kk = 0; kk < DMOD - 32; kk += 32) {
    ldA(kk + 32, araw[pb ^ 1]);
    ldB(kk + 32, braw[pb ^ 1]);
    compute(araw[pb], braw[pb]);
    pb ^= 1;
  }
  compute(araw[pb], braw[pb]);

#pragma unroll
  for (int nt = 0; nt < 2; ++nt) {
    const int sg = s0 + nt * 16 + ln;
    const int bb = sg >> 10, ss = sg & (SLEN - 1);
#pragma unroll
    for (int mt = 0; mt < 4; ++mt) {
      const int c = c0 + mt * 16 + lh * 8;
      const int hh = c >> 6, dd = c & 63;
      const unsigned oaddr =
          ((unsigned)(bb * NHEAD + hh) * SLEN + ss) * DHEAD + dd;
      if (MODE == 0) {
        v8bf r1, r2;
#pragma unroll
        for (int i = 0; i < 8; ++i) {
          const float base = acc[mt][nt][i] + bias[c + i];
          r1[i] = (__bf16)(base + ub[c + i]);
          r2[i] = (__bf16)(base + vb[c + i]);
        }
        *(v8bf*)(out1 + oaddr) = r1;
        *(v8bf*)(out2 + oaddr) = r2;
      } else {
        v8bf r;
#pragma unroll
        for (int i = 0; i < 8; ++i) {
          float v = acc[mt][nt][i];
          if (bias) v += bias[c + i];
          r[i] = (__bf16)v;
        }
        *(v8bf*)(out1 + oaddr) = r;
      }
    }
  }
}

// ---------------------------------------------------------------------------
// K1b: V projection, orientation M=s, N=c so D writes land contiguously in
// the transposed per-head layout vT[(b,h)][d][t].  Double-buffered K loop.
// ---------------------------------------------------------------------------
__global__ __launch_bounds__(256) void proj_v_kernel(
    const float* __restrict__ X, const __bf16* __restrict__ WT,
    const float* __restrict__ bias, __bf16* __restrict__ outT) {
  const int lane = threadIdx.x & 31, wave = threadIdx.x >> 5;
  const int ln = lane & 15, lh = lane >> 4;
  const int s0 = blockIdx.x * 128 + (wave & 1) * 64;   // M
  const int c0 = blockIdx.y * 128 + (wave >> 1) * 32;  // N

  unsigned arow[4], brow[2];
#pragma unroll
  for (int mt = 0; mt < 4; ++mt)
    arow[mt] = (unsigned)(s0 + mt * 16 + ln) * DMOD + lh * 16;
#pragma unroll
  for (int nt = 0; nt < 2; ++nt)
    brow[nt] = (unsigned)(c0 + nt * 16 + ln) * DMOD + lh * 16;

  v8f acc[4][2];
#pragma unroll
  for (int mt = 0; mt < 4; ++mt)
#pragma unroll
    for (int nt = 0; nt < 2; ++nt) acc[mt][nt] = vzero8();

  auto ldA = [&](unsigned kk, f32x4 (&dst)[4][4]) {
#pragma unroll
    for (int mt = 0; mt < 4; ++mt) {
      const float* p = X + (arow[mt] + kk);
      dst[mt][0] = *(const f32x4*)p;
      dst[mt][1] = *(const f32x4*)(p + 4);
      dst[mt][2] = *(const f32x4*)(p + 8);
      dst[mt][3] = *(const f32x4*)(p + 12);
    }
  };
  auto ldB = [&](unsigned kk, v16bf (&dst)[2]) {
#pragma unroll
    for (int nt = 0; nt < 2; ++nt)
      dst[nt] = *(const v16bf*)(WT + (brow[nt] + kk));
  };
  auto compute = [&](f32x4 (&ar)[4][4], v16bf (&br)[2]) {
    v16bf a[4];
#pragma unroll
    for (int mt = 0; mt < 4; ++mt) a[mt] = cvt16(ar[mt]);
#pragma unroll
    for (int mt = 0; mt < 4; ++mt)
#pragma unroll
      for (int nt = 0; nt < 2; ++nt)
        acc[mt][nt] = wmma_bf16(a[mt], br[nt], acc[mt][nt]);
  };

  f32x4 araw[2][4][4];
  v16bf braw[2][2];
  ldA(0, araw[0]);
  ldB(0, braw[0]);
  int pb = 0;
#pragma unroll 2
  for (unsigned kk = 0; kk < DMOD - 32; kk += 32) {
    ldA(kk + 32, araw[pb ^ 1]);
    ldB(kk + 32, braw[pb ^ 1]);
    compute(araw[pb], braw[pb]);
    pb ^= 1;
  }
  compute(araw[pb], braw[pb]);

#pragma unroll
  for (int nt = 0; nt < 2; ++nt) {
    const int c = c0 + nt * 16 + ln;
    const int hh = c >> 6, dd = c & 63;
    const float bc = bias[c];
#pragma unroll
    for (int mt = 0; mt < 4; ++mt) {
      const int sg = s0 + mt * 16 + lh * 8;
      const int bb = sg >> 10, ss = sg & (SLEN - 1);
      v8bf r;
#pragma unroll
      for (int i = 0; i < 8; ++i) r[i] = (__bf16)(acc[mt][nt][i] + bc);
      *(v8bf*)(outT +
               (((unsigned)(bb * NHEAD + hh) * DHEAD + dd) * SLEN + ss)) = r;
    }
  }
}

// ---------------------------------------------------------------------------
// K2: PS2[bh][s][u] = (q+v_bias)[s][:] . p[u][:]   (M=u, N=s, K=64)
// Both K-steps loaded up front, then 16 back-to-back WMMAs.
// ---------------------------------------------------------------------------
__global__ __launch_bounds__(256) void ps2_kernel(const __bf16* __restrict__ qv,
                                                  const __bf16* __restrict__ p,
                                                  __bf16* __restrict__ ps2) {
  const int lane = threadIdx.x & 31, wave = threadIdx.x >> 5;
  const int ln = lane & 15, lh = lane >> 4;
  const int bh = blockIdx.z;
  const int u0 = blockIdx.x * 128 + (wave & 1) * 64;
  const int s0 = blockIdx.y * 128 + (wave >> 1) * 32;
  const __bf16* pb = p + (unsigned)bh * SLEN * DHEAD;
  const __bf16* qb = qv + (unsigned)bh * SLEN * DHEAD;

  // load everything first (batched clauses)
  v8bf araw[2][4][2];
  v16bf braw[2][2];
#pragma unroll
  for (int kd = 0; kd < 2; ++kd) {
#pragma unroll
    for (int mt = 0; mt < 4; ++mt) {
      const __bf16* ap =
          pb + ((unsigned)(u0 + mt * 16 + ln) * DHEAD + kd * 32 + lh * 16);
      araw[kd][mt][0] = *(const v8bf*)ap;
      araw[kd][mt][1] = *(const v8bf*)(ap + 8);
    }
#pragma unroll
    for (int nt = 0; nt < 2; ++nt)
      braw[kd][nt] = *(const v16bf*)(
          qb + ((unsigned)(s0 + nt * 16 + ln) * DHEAD + kd * 32 + lh * 16));
  }

  v8f acc[4][2];
#pragma unroll
  for (int mt = 0; mt < 4; ++mt)
#pragma unroll
    for (int nt = 0; nt < 2; ++nt) acc[mt][nt] = vzero8();

#pragma unroll
  for (int kd = 0; kd < 2; ++kd)
#pragma unroll
    for (int mt = 0; mt < 4; ++mt) {
      const v16bf a = mk16(araw[kd][mt][0], araw[kd][mt][1]);
#pragma unroll
      for (int nt = 0; nt < 2; ++nt)
        acc[mt][nt] = wmma_bf16(a, braw[kd][nt], acc[mt][nt]);
    }

#pragma unroll
  for (int nt = 0; nt < 2; ++nt) {
    const int sg = s0 + nt * 16 + ln;
#pragma unroll
    for (int mt = 0; mt < 4; ++mt) {
      const int u = u0 + mt * 16 + lh * 8;
      v8bf r;
#pragma unroll
      for (int i = 0; i < 8; ++i) r[i] = (__bf16)acc[mt][nt][i];
      *(v8bf*)(ps2 + (((unsigned)bh * SLEN + sg) * SLEN + u)) = r;
    }
  }
}

// ---------------------------------------------------------------------------
// K3: fused flash attention with relative shift.
// Scores computed transposed (M=t, N=s): softmax row s == lane&15, so the
// running max/sum are per-lane scalars; D->B relayout of attn weights is done
// with shfl_xor only.  Tile loads (K rows, V rows, shifted PS2 values) are
// double-buffered across t-tiles; the rel-shift gather is branchless.
// ---------------------------------------------------------------------------
__global__ __launch_bounds__(256) void attn_kernel(
    const __bf16* __restrict__ qu, const __bf16* __restrict__ kM,
    const __bf16* __restrict__ vT, const __bf16* __restrict__ ps2,
    __bf16* __restrict__ ctx) {
  const int lane = threadIdx.x & 31, wave = threadIdx.x >> 5;
  const int ln = lane & 15, lh = lane >> 4;
  const int bh = blockIdx.y;
  const int b = bh >> 4, h = bh & 15;
  const int s = blockIdx.x * 128 + wave * 16 + ln;   // this lane's query row

  // Loop-invariant QU B-fragments (K = d, two 32-wide K-steps)
  const __bf16* qrow = qu + ((unsigned)(bh * SLEN + s) * DHEAD);
  v16bf qbf[2];
  qbf[0] = *(const v16bf*)(qrow + lh * 16);
  qbf[1] = *(const v16bf*)(qrow + 32 + lh * 16);

  const __bf16* kbase = kM + (unsigned)bh * SLEN * DHEAD;
  const __bf16* vbase = vT + (unsigned)bh * DHEAD * SLEN;
  const __bf16* psrow = ps2 + (unsigned)(bh * SLEN + s) * SLEN;

  float mrun = -3.0e38f, lrun = 0.0f;
  v8f cacc[4];
#pragma unroll
  for (int mt = 0; mt < 4; ++mt) cacc[mt] = vzero8();

  auto ldTile = [&](int t0, v8bf (&kr)[2][2][2], v8bf (&vr)[4][2],
                    float (&pos)[16]) {
#pragma unroll
    for (int mt = 0; mt < 2; ++mt)
#pragma unroll
      for (int kd = 0; kd < 2; ++kd) {
        const __bf16* ap = kbase + ((unsigned)(t0 + mt * 16 + ln) * DHEAD +
                                    kd * 32 + lh * 16);
        kr[mt][kd][0] = *(const v8bf*)ap;
        kr[mt][kd][1] = *(const v8bf*)(ap + 8);
      }
#pragma unroll
    for (int mt = 0; mt < 4; ++mt) {
      const __bf16* ap =
          vbase + ((unsigned)(mt * 16 + ln) * SLEN + t0 + lh * 16);
      vr[mt][0] = *(const v8bf*)ap;
      vr[mt][1] = *(const v8bf*)(ap + 8);
    }
    // branchless relative-shift gather:
    //   vcol <  S : PS2[s][vcol];  vcol == S : 0;  vcol > S : PS2[s+1][vcol-S-1]
    // psrow is row-s base, so rows s and s+1 are one contiguous span.
#pragma unroll
    for (int mt = 0; mt < 2; ++mt)
#pragma unroll
      for (int i = 0; i < 8; ++i) {
        const int t = t0 + mt * 16 + lh * 8 + i;
        const int vcol = t - s + (SLEN - 1);
        const int idx = vcol - (int)(vcol > SLEN);
        float pv = (float)psrow[(unsigned)idx];
        pos[mt * 8 + i] = (vcol == SLEN) ? 0.0f : pv;
      }
    __builtin_prefetch(kbase + (unsigned)(t0 + 64 + ln) * DHEAD, 0, 0);
  };

  auto computeTile = [&](v8bf (&kr)[2][2][2], v8bf (&vr)[4][2],
                         float (&pos)[16]) {
    // ---- content scores: score^T tile = K(t x d) x QU^T(d x s) ----
    v8f sc[2] = {vzero8(), vzero8()};
#pragma unroll
    for (int mt = 0; mt < 2; ++mt)
#pragma unroll
      for (int kd = 0; kd < 2; ++kd)
        sc[mt] = wmma_bf16(mk16(kr[mt][kd][0], kr[mt][kd][1]), qbf[kd], sc[mt]);

    // ---- add shifted positional scores, scale, track max ----
    float x[16];
    float tmax = -3.0e38f;
#pragma unroll
    for (int j = 0; j < 16; ++j) {
      const float xx = (sc[j >> 3][j & 7] + pos[j]) * 0.03125f;  // 1/sqrt(1024)
      x[j] = xx;
      tmax = fmaxf(tmax, xx);
    }
    tmax = fmaxf(tmax, __shfl_xor(tmax, 16, 32));

    // ---- online softmax (row state is a per-lane scalar) ----
    const float mnew = fmaxf(mrun, tmax);
    const float rescale = __expf(mrun - mnew);
    float w[16], wsum = 0.0f;
#pragma unroll
    for (int j = 0; j < 16; ++j) {
      w[j] = __expf(x[j] - mnew);
      wsum += w[j];
    }
    wsum += __shfl_xor(wsum, 16, 32);
    lrun = lrun * rescale + wsum;
    mrun = mnew;
#pragma unroll
    for (int mt = 0; mt < 4; ++mt)
#pragma unroll
      for (int i = 0; i < 8; ++i) cacc[mt][i] *= rescale;

    // ---- relay attn weights (D layout) -> B-fragment (t x s) via shuffles --
    v16bf battn;
#pragma unroll
    for (int j = 0; j < 8; ++j) {
      const float o0 = __shfl_xor(w[j], 16, 32);      // partner's mt0
      const float o1 = __shfl_xor(w[8 + j], 16, 32);  // partner's mt1
      battn[j]     = (__bf16)(lh == 0 ? w[j] : o1);
      battn[8 + j] = (__bf16)(lh == 0 ? o0 : w[8 + j]);
    }

    // ---- context^T accumulate: V^T(d x t) x attn(t x s) ----
#pragma unroll
    for (int mt = 0; mt < 4; ++mt)
      cacc[mt] = wmma_bf16(mk16(vr[mt][0], vr[mt][1]), battn, cacc[mt]);
  };

  v8bf kr[2][2][2][2];
  v8bf vr[2][4][2];
  float pos[2][16];
  ldTile(0, kr[0], vr[0], pos[0]);
  int pb = 0;
#pragma unroll 2
  for (int t0 = 0; t0 < SLEN - 32; t0 += 32) {
    ldTile(t0 + 32, kr[pb ^ 1], vr[pb ^ 1], pos[pb ^ 1]);
    computeTile(kr[pb], vr[pb], pos[pb]);
    pb ^= 1;
  }
  computeTile(kr[pb], vr[pb], pos[pb]);

  // ---- finalize and store context in token-major (b,s,h,d) bf16 ----
  const float inv = 1.0f / lrun;
  const unsigned obase = (unsigned)(b * SLEN + s) * DMOD + h * DHEAD;
#pragma unroll
  for (int mt = 0; mt < 4; ++mt) {
    v8bf r;
#pragma unroll
    for (int i = 0; i < 8; ++i) r[i] = (__bf16)(cacc[mt][i] * inv);
    *(v8bf*)(ctx + obase + mt * 16 + lh * 8) = r;
  }
}

// ---------------------------------------------------------------------------
// K4: output projection  out[s][c] = ctx[s][:] . WoT[c][:] + bo  (f32 out)
// Double-buffered K loop, all-bf16 operands.
// ---------------------------------------------------------------------------
__global__ __launch_bounds__(256) void outproj_kernel(
    const __bf16* __restrict__ ctx, const __bf16* __restrict__ WoT,
    const float* __restrict__ bo, float* __restrict__ out) {
  const int lane = threadIdx.x & 31, wave = threadIdx.x >> 5;
  const int ln = lane & 15, lh = lane >> 4;
  const int c0 = blockIdx.x * 128 + (wave & 1) * 64;
  const int s0 = blockIdx.y * 128 + (wave >> 1) * 32;

  unsigned arow[4], brow[2];
#pragma unroll
  for (int mt = 0; mt < 4; ++mt)
    arow[mt] = (unsigned)(c0 + mt * 16 + ln) * DMOD + lh * 16;
#pragma unroll
  for (int nt = 0; nt < 2; ++nt)
    brow[nt] = (unsigned)(s0 + nt * 16 + ln) * DMOD + lh * 16;

  v8f acc[4][2];
#pragma unroll
  for (int mt = 0; mt < 4; ++mt)
#pragma unroll
    for (int nt = 0; nt < 2; ++nt) acc[mt][nt] = vzero8();

  auto ldA = [&](unsigned kk, v8bf (&dst)[4][2]) {
#pragma unroll
    for (int mt = 0; mt < 4; ++mt) {
      const __bf16* p = WoT + (arow[mt] + kk);
      dst[mt][0] = *(const v8bf*)p;
      dst[mt][1] = *(const v8bf*)(p + 8);
    }
  };
  auto ldB = [&](unsigned kk, v16bf (&dst)[2]) {
#pragma unroll
    for (int nt = 0; nt < 2; ++nt)
      dst[nt] = *(const v16bf*)(ctx + (brow[nt] + kk));
  };
  auto compute = [&](v8bf (&ar)[4][2], v16bf (&br)[2]) {
#pragma unroll
    for (int mt = 0; mt < 4; ++mt) {
      const v16bf a = mk16(ar[mt][0], ar[mt][1]);
#pragma unroll
      for (int nt = 0; nt < 2; ++nt)
        acc[mt][nt] = wmma_bf16(a, br[nt], acc[mt][nt]);
    }
  };

  v8bf araw[2][4][2];
  v16bf braw[2][2];
  ldA(0, araw[0]);
  ldB(0, braw[0]);
  int pb = 0;
#pragma unroll 2
  for (unsigned kk = 0; kk < DMOD - 32; kk += 32) {
    ldA(kk + 32, araw[pb ^ 1]);
    ldB(kk + 32, braw[pb ^ 1]);
    compute(araw[pb], braw[pb]);
    pb ^= 1;
  }
  compute(araw[pb], braw[pb]);

#pragma unroll
  for (int nt = 0; nt < 2; ++nt) {
    const int sg = s0 + nt * 16 + ln;
#pragma unroll
    for (int mt = 0; mt < 4; ++mt) {
      const int c = c0 + mt * 16 + lh * 8;
      v8f r;
#pragma unroll
      for (int i = 0; i < 8; ++i) r[i] = acc[mt][nt][i] + bo[c + i];
      *(v8f*)(out + ((unsigned)sg * DMOD + c)) = r;
    }
  }
}

// ---------------------------------------------------------------------------
extern "C" void kernel_launch(void* const* d_in, const int* in_sizes, int n_in,
                              void* d_out, int out_size, void* d_ws,
                              size_t ws_size, hipStream_t stream) {
  (void)in_sizes; (void)n_in; (void)out_size; (void)ws_size;
  const float* query = (const float*)d_in[0];
  const float* key_  = (const float*)d_in[1];
  const float* value = (const float*)d_in[2];
  const float* pose  = (const float*)d_in[3];
  const float* Wq = (const float*)d_in[4];
  const float* bq = (const float*)d_in[5];
  const float* Wk = (const float*)d_in[6];
  const float* bk = (const float*)d_in[7];
  const float* Wv = (const float*)d_in[8];
  const float* bv = (const float*)d_in[9];
  const float* Wp = (const float*)d_in[10];
  const float* ub = (const float*)d_in[11];
  const float* vb = (const float*)d_in[12];
  const float* Wo = (const float*)d_in[13];
  const float* bo = (const float*)d_in[14];

  char* ws = (char*)d_ws;
  __bf16* wqT = (__bf16*)(ws + OFF_WQT);
  __bf16* wkT = (__bf16*)(ws + OFF_WKT);
  __bf16* wvT = (__bf16*)(ws + OFF_WVT);
  __bf16* wpT = (__bf16*)(ws + OFF_WPT);
  __bf16* woT = (__bf16*)(ws + OFF_WOT);
  __bf16* quB = (__bf16*)(ws + OFF_QU);
  __bf16* qvB = (__bf16*)(ws + OFF_QV);
  __bf16* kB  = (__bf16*)(ws + OFF_K);
  __bf16* vTB = (__bf16*)(ws + OFF_VT);
  __bf16* pB  = (__bf16*)(ws + OFF_P);
  __bf16* psB = (__bf16*)(ws + OFF_PS2);
  __bf16* cxB = (__bf16*)(ws + OFF_CTX);
  float* out = (float*)d_out;

  const dim3 tTB(32, 8);
  wprep_kernel<<<dim3(32, 32), tTB, 0, stream>>>(Wq, wqT);
  wprep_kernel<<<dim3(32, 32), tTB, 0, stream>>>(Wk, wkT);
  wprep_kernel<<<dim3(32, 32), tTB, 0, stream>>>(Wv, wvT);
  wprep_kernel<<<dim3(32, 32), tTB, 0, stream>>>(Wp, wpT);
  wprep_kernel<<<dim3(32, 32), tTB, 0, stream>>>(Wo, woT);

  proj_cs_kernel<0><<<dim3(8, 32), 256, 0, stream>>>(query, wqT, bq, ub, vb, quB, qvB);
  proj_cs_kernel<1><<<dim3(8, 32), 256, 0, stream>>>(key_, wkT, bk, nullptr, nullptr, kB, nullptr);
  proj_v_kernel<<<dim3(32, 8), 256, 0, stream>>>(value, wvT, bv, vTB);
  proj_cs_kernel<3><<<dim3(8, 32), 256, 0, stream>>>(pose, wpT, nullptr, nullptr, nullptr, pB, nullptr);

  ps2_kernel<<<dim3(8, 8, BATCH * NHEAD), 256, 0, stream>>>(qvB, pB, psB);
  attn_kernel<<<dim3(8, BATCH * NHEAD), 256, 0, stream>>>(quB, kB, vTB, psB, cxB);
  outproj_kernel<<<dim3(8, 32), 256, 0, stream>>>(cxB, woT, bo, out);
}
